// FusedMoEMethod_73100343378287
// MI455X (gfx1250) — compile-verified, matrix operand order checked by reading, not run on hardware
//
#include <hip/hip_runtime.h>
#include <hip/hip_bf16.h>

// ---------------------------------------------------------------------------
// Fused MoE for MI455X (gfx1250, wave32, WMMA bf16).
//   kernel 0: zero d_out + per-expert counters
//   kernel 1: router softmax -> top2 -> renorm -> scatter token lists
//   kernel 2: stream-convert w13/w2 fp32 -> bf16 into d_ws (one pass)
//   kernel 3: per (expert, 48-token tile) fused SiLU-MLP, 16 waves:
//             GEMM1 (v_wmma_f32_16x16x32_bf16) -> silu*up -> GEMM2 ->
//             weighted atomic scatter into d_out.
// Weight traffic scales as 1/TILE_M (17.3 MB bf16 per block, no duplication
// across waves). TILE_M=48 -> ~342 blocks -> ~5.9 GB L2 traffic, with
// 120.5 KB LDS (safe under a 160 KB CU-mode cap) and ~190 VGPRs.
// ---------------------------------------------------------------------------

typedef __attribute__((ext_vector_type(16))) __bf16 bf16x16;
typedef __attribute__((ext_vector_type(4)))  __bf16 bf16x4;
typedef __attribute__((ext_vector_type(8)))  float  floatx8;

namespace {
constexpr int kT = 8192;     // tokens
constexpr int kH = 1024;     // hidden
constexpr int kI = 2816;     // intermediate
constexpr int kE = 8;        // experts
constexpr int TILE_M = 48;   // token tile (3 WMMA M-tiles)
constexpr int MT     = 3;    // M-tiles per block
constexpr int NWAVE  = 16;   // waves per block
constexpr int NTHR   = 512;  // threads per block
constexpr int IC     = 256;  // I-chunk = NWAVE*16 ; kI = 11*IC exactly
constexpr int HT     = 4;    // H-tiles per wave (16 waves * 4 * 16 = 1024)
}

__device__ __forceinline__ __bf16 f32_to_bf16(float f) {
  return (__bf16)f;          // native bf16 cvt on gfx1250
}

// ---- WMMA operand fragment loaders (16 contiguous K elements per lane) ----
__device__ __forceinline__ bf16x16 load_frag(const __bf16* __restrict__ p) {
  return *(const bf16x16*)p;                 // 2x global_load_b128 / ds_read
}
__device__ __forceinline__ bf16x16 load_frag(const float* __restrict__ p) {
  bf16x16 r;                                 // fallback: convert on load
#pragma unroll
  for (int j = 0; j < 16; j += 4) {
    float4 f = *(const float4*)(p + j);
    r[j + 0] = f32_to_bf16(f.x);
    r[j + 1] = f32_to_bf16(f.y);
    r[j + 2] = f32_to_bf16(f.z);
    r[j + 3] = f32_to_bf16(f.w);
  }
  return r;
}

__device__ __forceinline__ floatx8 wmma_bf16(bf16x16 a, bf16x16 b, floatx8 c) {
  return __builtin_amdgcn_wmma_f32_16x16x32_bf16(
      /*neg_a=*/false, a, /*neg_b=*/false, b,
      /*c_mod=*/(short)0, c, /*reuse_a=*/false, /*reuse_b=*/false);
}

// --------------------------- kernel 0: zero-init ---------------------------
__global__ void moe_zero_kernel(float* __restrict__ out, size_t n,
                                int* __restrict__ cnt) {
  size_t i = (size_t)blockIdx.x * blockDim.x + threadIdx.x;
  if (i < (size_t)kE) cnt[i] = 0;
  size_t stride = (size_t)gridDim.x * blockDim.x;
  for (size_t j = i; j < n; j += stride) out[j] = 0.0f;
}

// --------------------------- kernel 1: router ------------------------------
__global__ void moe_router_kernel(const float* __restrict__ logits,
                                  int* __restrict__ cnt,
                                  int* __restrict__ ids,
                                  float* __restrict__ wts) {
  int t = blockIdx.x * blockDim.x + threadIdx.x;
  if (t >= kT) return;
  float l[kE];
  float m = -3.4e38f;
#pragma unroll
  for (int e = 0; e < kE; ++e) { l[e] = logits[t * kE + e]; m = fmaxf(m, l[e]); }
#pragma unroll
  for (int e = 0; e < kE; ++e) l[e] = __expf(l[e] - m);   // softmax numerators
  int i0 = 0, i1 = -1; float p0 = l[0], p1 = -1.0f;
#pragma unroll
  for (int e = 1; e < kE; ++e) {
    float p = l[e];
    if (p > p0) { p1 = p0; i1 = i0; p0 = p; i0 = e; }
    else if (p > p1) { p1 = p; i1 = e; }
  }
  float inv = 1.0f / (p0 + p1);               // denominator cancels in renorm
  int s0 = atomicAdd(&cnt[i0], 1);
  ids[i0 * kT + s0] = t;  wts[i0 * kT + s0] = p0 * inv;
  int s1 = atomicAdd(&cnt[i1], 1);
  ids[i1 * kT + s1] = t;  wts[i1 * kT + s1] = p1 * inv;
}

// -------------- kernel 2: streaming fp32 -> bf16 weight convert ------------
__global__ void moe_cvt_kernel(const float* __restrict__ src,
                               __bf16* __restrict__ dst, size_t n) {
  size_t i = ((size_t)blockIdx.x * blockDim.x + threadIdx.x) * 4;
  size_t stride = (size_t)gridDim.x * blockDim.x * 4;
  for (; i + 3 < n; i += stride) {
    float4 f = *(const float4*)(src + i);
    bf16x4 v = { f32_to_bf16(f.x), f32_to_bf16(f.y),
                 f32_to_bf16(f.z), f32_to_bf16(f.w) };
    *(bf16x4*)(dst + i) = v;                   // 8-byte packed store
  }
}

// ------------------ kernel 3: fused expert MLP (WMMA bf16) -----------------
// grid = (ceil(kT/TILE_M), kE), block = 512 (16 waves). Blocks past the
// routed token count exit immediately. WT = __bf16 (pre-converted weights)
// or float (inline-convert fallback when d_ws is too small).
template <typename WT>
__global__ __launch_bounds__(NTHR)
void moe_expert_kernel(const float* __restrict__ x,     // [T,H] fp32
                       const WT*    __restrict__ w13,   // [E,2I,H]
                       const WT*    __restrict__ w2,    // [E,H,I]
                       const int*   __restrict__ cnt,   // [E]
                       const int*   __restrict__ ids,   // [E,T]
                       const float* __restrict__ wts,   // [E,T]
                       float* __restrict__ out) {       // [T,H]
  __shared__ __bf16 xs[TILE_M][kH];   // 96 KB: gathered x tile (bf16)
  __shared__ __bf16 hs[TILE_M][IC];   // 24 KB: silu(gate)*up chunk
  __shared__ int    tok[TILE_M];
  __shared__ float  tw[TILE_M];

  const int e    = blockIdx.y;
  const int tile = blockIdx.x;
  const int n_e  = cnt[e];
  if (tile * TILE_M >= n_e) return;

  const int tid   = threadIdx.x;
  const int lane  = tid & 31;
  const int wave  = tid >> 5;    // 0..15
  const int col   = lane & 15;   // N (or M for A-frag) index within 16
  const int khalf = lane >> 4;   // which 16-wide K half this lane holds

  // ---- token ids / combine weights for this tile (pad with weight 0) ----
  if (tid < TILE_M) {
    int slot = tile * TILE_M + tid;
    if (slot < n_e) { tok[tid] = ids[e * kT + slot]; tw[tid] = wts[e * kT + slot]; }
    else            { tok[tid] = ids[e * kT + tile * TILE_M]; tw[tid] = 0.0f; }
  }
  __syncthreads();

  // ---- gather x rows -> LDS as bf16 (48 rows x 1024 cols, 512 threads) ----
  {
    int c0 = (tid & 15) * 64;     // 64 cols per thread, 16 threads per row
    for (int r = tid >> 4; r < TILE_M; r += NTHR / 16) {
      const float* xr = x + (size_t)tok[r] * kH + c0;
#pragma unroll
      for (int c = 0; c < 64; c += 4) {
        float4 f = *(const float4*)(xr + c);
        xs[r][c0 + c + 0] = f32_to_bf16(f.x);
        xs[r][c0 + c + 1] = f32_to_bf16(f.y);
        xs[r][c0 + c + 2] = f32_to_bf16(f.z);
        xs[r][c0 + c + 3] = f32_to_bf16(f.w);
      }
    }
  }
  __syncthreads();

  const WT* w13e = w13 + (size_t)e * (2 * kI) * kH;
  const WT* w2e  = w2  + (size_t)e * kH * kI;

  // per-wave W2 row base pointers (4 H-tiles per wave)
  const WT* w2row[HT];
#pragma unroll
  for (int t = 0; t < HT; ++t)
    w2row[t] = w2e + (size_t)((wave * HT + t) * 16 + col) * kI;

  // out accumulators: MT x HT 16x16 tiles per wave (covers [48, 1024])
  floatx8 acc2[HT][MT];
#pragma unroll
  for (int t = 0; t < HT; ++t)
#pragma unroll
    for (int mt = 0; mt < MT; ++mt) acc2[t][mt] = (floatx8){0,0,0,0,0,0,0,0};

  for (int i0 = 0; i0 < kI; i0 += IC) {
    // ================= GEMM1: gate/up for this wave's 16 i-cols ==========
    const int irow = i0 + wave * 16 + col;                 // I-dim column
    const WT* bg   = w13e + (size_t)irow * kH;             // W1 row (gate)
    const WT* bu   = w13e + (size_t)(kI + irow) * kH;      // W3 row (up)

    floatx8 gacc[MT], uacc[MT];
#pragma unroll
    for (int mt = 0; mt < MT; ++mt) {
      gacc[mt] = (floatx8){0,0,0,0,0,0,0,0};
      uacc[mt] = (floatx8){0,0,0,0,0,0,0,0};
    }
    for (int k0 = 0; k0 < kH; k0 += 32) {
      int ka = k0 + khalf * 16;
      __builtin_prefetch(bg + ka + 128, 0, 1);   // speculative; OOB dropped
      __builtin_prefetch(bu + ka + 128, 0, 1);
      bf16x16 b1 = load_frag(bg + ka);                     // B: W1 col strip
      bf16x16 b3 = load_frag(bu + ka);                     // B: W3 col strip
#pragma unroll
      for (int mt = 0; mt < MT; ++mt) {
        bf16x16 a = *(const bf16x16*)&xs[mt * 16 + col][ka]; // A: x (LDS)
        gacc[mt] = wmma_bf16(a, b1, gacc[mt]);             // B reused MT x
        uacc[mt] = wmma_bf16(a, b3, uacc[mt]);
      }
    }

    // ---- h = silu(gate) * up -> LDS chunk [48 x IC] (bf16) ----
#pragma unroll
    for (int mt = 0; mt < MT; ++mt)
#pragma unroll
      for (int j = 0; j < 8; ++j) {
        float g = gacc[mt][j];
        float u = uacc[mt][j];
        float h = g * __builtin_amdgcn_rcpf(1.0f + __expf(-g)) * u;
        int m = mt * 16 + j + khalf * 8;                   // C/D VGPR layout
        hs[m][wave * 16 + col] = f32_to_bf16(h);
      }
    __syncthreads();

    // ================= GEMM2: out[48,1024] += h_chunk @ W2_chunk^T ========
    for (int kk = 0; kk < IC; kk += 32) {
      const int koff = i0 + kk + khalf * 16;
      bf16x16 a2[MT];
#pragma unroll
      for (int mt = 0; mt < MT; ++mt)
        a2[mt] = *(const bf16x16*)&hs[mt * 16 + col][kk + khalf * 16];
#pragma unroll
      for (int t = 0; t < HT; ++t) {
        __builtin_prefetch(w2row[t] + koff + 256, 0, 1);
        bf16x16 b2 = load_frag(w2row[t] + koff);
#pragma unroll
        for (int mt = 0; mt < MT; ++mt)
          acc2[t][mt] = wmma_bf16(a2[mt], b2, acc2[t][mt]); // B reused 3x
      }
    }
    __syncthreads();   // protect hs before next chunk overwrites it
  }

  // ---- epilogue: weighted scatter (exactly 2 atomic adds per out element,
  //      fp add commutative -> bitwise deterministic) ----
#pragma unroll
  for (int t = 0; t < HT; ++t) {
    int n0 = (wave * HT + t) * 16;
#pragma unroll
    for (int mt = 0; mt < MT; ++mt)
#pragma unroll
      for (int j = 0; j < 8; ++j) {
        int m = mt * 16 + j + khalf * 8;
        float v = acc2[t][mt][j] * tw[m];
        atomicAdd(&out[(size_t)tok[m] * kH + n0 + col], v);
      }
  }
}

// ------------------------------ launcher -----------------------------------
extern "C" void kernel_launch(void* const* d_in, const int* in_sizes, int n_in,
                              void* d_out, int out_size, void* d_ws, size_t ws_size,
                              hipStream_t stream) {
  const float* x      = (const float*)d_in[0];   // [T,H]
  const float* logits = (const float*)d_in[1];   // [T,E]
  const float* w13    = (const float*)d_in[2];   // [E,2I,H]
  const float* w2     = (const float*)d_in[3];   // [E,H,I]
  // d_in[4] = top_k (== 2, fixed by the reference shapes)
  float* out = (float*)d_out;

  // workspace layout
  constexpr size_t W13N    = (size_t)kE * 2 * kI * kH;   // 46.1M elems
  constexpr size_t W2N     = (size_t)kE * kH * kI;       // 23.1M elems
  constexpr size_t OFF_IDS = 1024;
  constexpr size_t OFF_WTS = OFF_IDS + (size_t)kE * kT * 4;
  constexpr size_t OFF_W13 = 1ull << 20;                 // 1 MB
  constexpr size_t OFF_W2  = OFF_W13 + W13N * 2;
  constexpr size_t NEED    = OFF_W2 + W2N * 2;           // ~139.5 MB

  char*  ws  = (char*)d_ws;
  int*   cnt = (int*)ws;
  int*   ids = (int*)(ws + OFF_IDS);
  float* wts = (float*)(ws + OFF_WTS);

  moe_zero_kernel<<<512, 256, 0, stream>>>(out, (size_t)kT * kH, cnt);
  moe_router_kernel<<<kT / 256, 256, 0, stream>>>(logits, cnt, ids, wts);

  dim3 grid((kT + TILE_M - 1) / TILE_M, kE);   // idle tiles exit immediately

  if (ws_size >= NEED) {
    // pre-convert weights once; GEMM loops become pure load+wmma
    __bf16* w13b = (__bf16*)(ws + OFF_W13);
    __bf16* w2b  = (__bf16*)(ws + OFF_W2);
    moe_cvt_kernel<<<2048, 256, 0, stream>>>(w13, w13b, W13N);
    moe_cvt_kernel<<<2048, 256, 0, stream>>>(w2,  w2b,  W2N);
    moe_expert_kernel<__bf16><<<grid, NTHR, 0, stream>>>(
        x, (const __bf16*)w13b, (const __bf16*)w2b, cnt, ids, wts, out);
  } else {
    // fallback: convert weights in-register on load
    moe_expert_kernel<float><<<grid, NTHR, 0, stream>>>(
        x, w13, w2, cnt, ids, wts, out);
  }
}
